// PromptVQ_60765197304295
// MI455X (gfx1250) — matrix operand order repaired and checked by reading, not compiled, not run on hardware
//
#include <hip/hip_runtime.h>

// ---------------------------------------------------------------------------
// VQ nearest-codebook search on MI455X (gfx1250).
//   argmin_k ||x - c_k||^2 == argmin_k ( ||c_k||^2 - 2 x.c_k )   (x-norm dropped)
// Distance GEMM (2*N*K*D = 412 GFLOP) via V_WMMA_F32_16X16X32_BF16 with a bf16
// hi/lo split (3 WMMAs / 16x16x32 tile -> ~fp32-accurate dots).
// Codebook is pre-split into bf16 hi/lo planes in d_ws; the main loop streams
// it into LDS with the Tensor Data Mover (tensor_load_to_lds, TENSORcnt),
// double-buffered against the WMMA phase with compile-time ping-pong buffers.
// ---------------------------------------------------------------------------

#define D_DIM   768
#define M_TILE  64       // rows of x per workgroup (4 row-groups of 16)
#define NT_STEP 32       // codes per code-step (2 code-groups of 16)
#define D_HALF  384      // D sliced in halves for TDM double buffering

typedef __attribute__((ext_vector_type(16))) __bf16 v16bf;
typedef __attribute__((ext_vector_type(8)))  __bf16 v8bf;
typedef __attribute__((ext_vector_type(8)))  float  v8f;
typedef __attribute__((ext_vector_type(4)))  unsigned int v4u;
typedef __attribute__((ext_vector_type(8)))  int    v8i;
typedef __attribute__((ext_vector_type(4)))  int    v4i;

// ---- c2[k] = ||codebook[k]||^2 : one wave32 per code ----------------------
__global__ void vq_c2_kernel(const float* __restrict__ cb,
                             float* __restrict__ c2, int K) {
  int gwave = (int)((blockIdx.x * blockDim.x + threadIdx.x) >> 5);
  int lane  = threadIdx.x & 31;
  if (gwave >= K) return;
  const float* row = cb + (size_t)gwave * D_DIM;
  float s = 0.0f;
  for (int d = lane; d < D_DIM; d += 32) { float v = row[d]; s += v * v; }
  #pragma unroll
  for (int off = 16; off > 0; off >>= 1) s += __shfl_xor(s, off, 32);
  if (lane == 0) c2[gwave] = s;
}

// ---- codebook fp32 -> bf16 hi/lo planes (done once, L2-resident source) ---
__global__ void vq_cvt_cb_kernel(const float* __restrict__ cb,
                                 __bf16* __restrict__ cbHi,
                                 __bf16* __restrict__ cbLo, int total) {
  int i = (int)(blockIdx.x * blockDim.x + threadIdx.x);
  if (i >= total) return;
  float f = cb[i];
  __bf16 hi = (__bf16)f;
  cbHi[i] = hi;
  cbLo[i] = (__bf16)(f - (float)hi);
}

// ---- TDM: one 3D tile [2 planes][NT_STEP rows][D_HALF cols] of bf16 -> LDS.
// Descriptor per CDNA5 ISA ch.8: group0 {count,lds,global,type}, group1
// {data_size, dims/strides}, group2 {dim2}, group3 zero.  Issued by wave 0
// only (wave-level DMA, EXEC ignored), tracked with TENSORcnt.
// clang-23 toolchain: 6-arg builtin, 5th arg (VADDR4 slot) zero-filled.
__device__ __forceinline__ void tdm_load_slice(const __bf16* gsrc,
                                               unsigned ldsOff,
                                               unsigned planeStrideElems) {
  unsigned long long ga = (unsigned long long)(uintptr_t)gsrc;
  v4u g0;
  g0[0] = 1u;                                       // count=1, user descriptor
  g0[1] = ldsOff;                                   // lds_addr (bytes)
  g0[2] = (unsigned)(ga & 0xffffffffu);             // global_addr[31:0]
  g0[3] = (unsigned)((ga >> 32) & 0x01ffffffu)      // global_addr[56:32]
        | (2u << 30);                               // type=2 ("image")
  v8i g1;
  g1[0] = (int)(1u << 16);                          // wg_mask=0; data_size=1 (2B)
  g1[1] = (int)((unsigned)D_HALF << 16);            // tensor_dim0[15:0]
  g1[2] = (int)((unsigned)NT_STEP << 16);           // tensor_dim0 hi=0; tensor_dim1[15:0]
  g1[3] = (int)((unsigned)D_HALF << 16);            // tensor_dim1 hi=0; tile_dim0
  g1[4] = (int)((unsigned)NT_STEP | (2u << 16));    // tile_dim1; tile_dim2=2 planes
  g1[5] = D_DIM;                                    // tensor_dim0_stride[31:0]
  g1[6] = (int)((planeStrideElems & 0xffffu) << 16);// d0_stride hi=0; d1_stride[15:0]
  g1[7] = (int)(planeStrideElems >> 16);            // d1_stride[47:16]
  v4i g2;
  g2[0] = 2;                                        // tensor_dim2 = 2 planes
  g2[1] = 0; g2[2] = 0; g2[3] = 0;                  // no dim3 / stride2 / tile_dim3
  v4i g3;
  g3[0] = 0; g3[1] = 0; g3[2] = 0; g3[3] = 0;
  v8i g4;                                           // unused VADDR4 slot: zeros
  g4[0] = 0; g4[1] = 0; g4[2] = 0; g4[3] = 0;
  g4[4] = 0; g4[5] = 0; g4[6] = 0; g4[7] = 0;
  __builtin_amdgcn_tensor_load_to_lds(g0, g1, g2, g3, g4, 0);
}

// ---- 36-WMMA compute phase on one staged D-half ---------------------------
__device__ __forceinline__ void compute_half(const __bf16* __restrict__ Ahi,
                                             const __bf16* __restrict__ Alo,
                                             const __bf16* __restrict__ Bbase,
                                             int aRow, int bCol, int h, int dbase,
                                             v8f& acc0, v8f& acc1) {
  const __bf16* Bh = Bbase + (0 * NT_STEP + bCol) * D_HALF;
  const __bf16* Bl = Bbase + (1 * NT_STEP + bCol) * D_HALF;
  #pragma unroll
  for (int c = 0; c < D_HALF / 32; ++c) {
    const int d0 = c * 32;
    // A fragment, k order chosen here (same as B): lane half h holds K = 16h..16h+15 (contiguous).
    v8bf a1 = *(const v8bf*)&Ahi[aRow * D_DIM + dbase + d0 + 16 * h];
    v8bf a2 = *(const v8bf*)&Ahi[aRow * D_DIM + dbase + d0 + 16 * h + 8];
    v8bf a3 = *(const v8bf*)&Alo[aRow * D_DIM + dbase + d0 + 16 * h];
    v8bf a4 = *(const v8bf*)&Alo[aRow * D_DIM + dbase + d0 + 16 * h + 8];
    v16bf ah, al;
    #pragma unroll
    for (int e = 0; e < 8; ++e) {
      ah[e] = a1[e]; ah[e + 8] = a2[e];
      al[e] = a3[e]; al[e + 8] = a4[e];
    }
    // B fragment, k order chosen here (same as A): lane half h holds K = 16h..16h+15 (contiguous).
    v16bf bh = *(const v16bf*)&Bh[d0 + 16 * h];
    v16bf bl = *(const v16bf*)&Bl[d0 + 16 * h];

    acc0 = __builtin_amdgcn_wmma_f32_16x16x32_bf16(false, ah, false, bh,
                                                   (short)0, acc0, false, false);
    acc1 = __builtin_amdgcn_wmma_f32_16x16x32_bf16(false, al, false, bh,
                                                   (short)0, acc1, false, false);
    acc1 = __builtin_amdgcn_wmma_f32_16x16x32_bf16(false, ah, false, bl,
                                                   (short)0, acc1, false, false);
  }
}

// ---- main fused GEMM + argmin + gather ------------------------------------
__global__ __launch_bounds__(256, 1)
void vq_argmin_kernel(const float* __restrict__ x,
                      const float* __restrict__ cb,
                      const __bf16* __restrict__ cbHi,   // [K][768], lo plane follows
                      const float* __restrict__ c2,
                      float* __restrict__ qout,
                      int* __restrict__ idx_out,
                      int K) {
  // LDS: A tile 64x768 bf16 hi/lo (192 KB) + 2 TDM B buffers (2x48 KB) = 288 KB.
  __shared__ __align__(32) __bf16 Ahi[M_TILE * D_DIM];
  __shared__ __align__(32) __bf16 Alo[M_TILE * D_DIM];
  __shared__ __align__(32) __bf16 Bbuf[2][2 * NT_STEP * D_HALF]; // [hi|lo][row][col]

  const int tid  = threadIdx.x;
  const int wave = tid >> 5;        // 0..7
  const int lane = tid & 31;
  const int rg   = wave & 3;        // row group (16 rows each)
  const int cg   = wave >> 2;       // code group (16 codes each)
  const int h    = lane >> 4;       // lane half
  const int n    = lane & 15;
  const size_t row_base = (size_t)blockIdx.x * M_TILE;

  const unsigned planeStride = (unsigned)K * D_DIM;  // hi->lo plane, in elements
  const unsigned bLds0 = (unsigned)(uintptr_t)(void*)&Bbuf[0][0];
  const unsigned bLds1 = (unsigned)(uintptr_t)(void*)&Bbuf[1][0];

  // Kick off TDM preload of (nt=0, dh=0) while we stage A (overlaps the cvt).
  if (wave == 0) tdm_load_slice(cbHi, bLds0, planeStride);

  // Stage x tile -> LDS as bf16 hi/lo (hi = RNE(x), lo = RNE(x - hi)).
  for (int i = tid; i < M_TILE * D_DIM; i += 256) {
    float f = x[row_base * D_DIM + i];
    __bf16 hi = (__bf16)f;
    Ahi[i] = hi;
    Alo[i] = (__bf16)(f - (float)hi);
  }

  if (wave == 0) __builtin_amdgcn_s_wait_tensorcnt(0);
  __syncthreads();                  // A staged AND Bbuf[0] ready

  float bestD[8];
  int   bestI[8];
  #pragma unroll
  for (int r = 0; r < 8; ++r) { bestD[r] = 3.4e38f; bestI[r] = 0; }

  const int aRow = rg * 16 + n;     // A-fragment: lane holds row M = lane%16
  const int bCol = cg * 16 + n;     // B-fragment: lane holds column N = lane%16

  // Fixed ping-pong: dh=0 computes Bbuf[0] / prefetches Bbuf[1]; dh=1 reverse.
  for (int nt = 0; nt < K; nt += NT_STEP) {
    v8f acc0 = {0.f,0.f,0.f,0.f,0.f,0.f,0.f,0.f};
    v8f acc1 = {0.f,0.f,0.f,0.f,0.f,0.f,0.f,0.f};

    // ---- half 0: compute on Bbuf[0]; DMA (nt, dh=1) into Bbuf[1] ----
    if (wave == 0)
      tdm_load_slice(cbHi + (size_t)nt * D_DIM + D_HALF, bLds1, planeStride);
    compute_half(Ahi, Alo, &Bbuf[0][0], aRow, bCol, h, 0, acc0, acc1);
    __syncthreads();                                     // done with Bbuf[0]
    if (wave == 0) __builtin_amdgcn_s_wait_tensorcnt(0); // Bbuf[1] landed
    __syncthreads();                                     // publish Bbuf[1]

    // ---- half 1: compute on Bbuf[1]; DMA (nt+32, dh=0) into Bbuf[0] ----
    if (wave == 0 && nt + NT_STEP < K)
      tdm_load_slice(cbHi + (size_t)(nt + NT_STEP) * D_DIM, bLds0, planeStride);
    compute_half(Ahi, Alo, &Bbuf[1][0], aRow, bCol, h, D_HALF, acc0, acc1);

    // C layout: VGPR r, lane L -> row (r + 8*(L/16)), col (L%16).
    const int  code = nt + cg * 16 + n;
    const float cc  = c2[code];
    #pragma unroll
    for (int r = 0; r < 8; ++r) {
      float dist = cc - 2.0f * (acc0[r] + acc1[r]);
      if (dist < bestD[r]) { bestD[r] = dist; bestI[r] = code; }
    }

    __syncthreads();                                     // done with Bbuf[1]
    if (wave == 0) __builtin_amdgcn_s_wait_tensorcnt(0); // Bbuf[0] landed
    __syncthreads();                                     // publish Bbuf[0]
  }

  // Cross-lane / cross-code-group argmin reduction. Reuse dead B LDS as scratch.
  float* redD = (float*)&Bbuf[0][0];          // [8 waves][16 rows][16 lanes]
  int*   redI = (int*)&Bbuf[1][0];
  int*   bestIdxLds = redI + 8 * 16 * 16;

  #pragma unroll
  for (int r = 0; r < 8; ++r) {
    int rl = r + 8 * h;
    redD[(wave * 16 + rl) * 16 + n] = bestD[r];
    redI[(wave * 16 + rl) * 16 + n] = bestI[r];
  }
  __syncthreads();

  if (tid < M_TILE) {                          // one thread per output row
    int rgt = tid >> 4, rl = tid & 15;
    float bd = 3.4e38f; int bi = 0;
    #pragma unroll
    for (int cgi = 0; cgi < 2; ++cgi) {
      int w = rgt + 4 * cgi;
      for (int nn = 0; nn < 16; ++nn) {
        float dv = redD[(w * 16 + rl) * 16 + nn];
        if (dv < bd) { bd = dv; bi = redI[(w * 16 + rl) * 16 + nn]; }
      }
    }
    bestIdxLds[tid] = bi;
    idx_out[row_base + tid] = bi;
  }
  __syncthreads();

  // Gather: qout[row] = codebook[best[row]] (fp32 forward value of the STE).
  for (int rr = 0; rr < M_TILE; ++rr) {
    const float* src = cb + (size_t)bestIdxLds[rr] * D_DIM;
    float* dst = qout + (row_base + rr) * D_DIM;
    for (int j = tid; j < D_DIM; j += 256) dst[j] = src[j];
  }
}

extern "C" void kernel_launch(void* const* d_in, const int* in_sizes, int n_in,
                              void* d_out, int out_size, void* d_ws, size_t ws_size,
                              hipStream_t stream) {
  const float* x  = (const float*)d_in[0];
  const float* cb = (const float*)d_in[1];
  const int N = in_sizes[0] / D_DIM;     // 65536
  const int K = in_sizes[1] / D_DIM;     // 4096

  float* qout    = (float*)d_out;                             // [N, D] fp32
  int*   idx_out = (int*)((float*)d_out + (size_t)N * D_DIM); // [N] int32, concat

  // Workspace layout: c2 (K floats, 16 KB) | cbHi (K*D bf16) | cbLo (K*D bf16).
  float*  c2   = (float*)d_ws;
  __bf16* cbHi = (__bf16*)((char*)d_ws + 16384);
  __bf16* cbLo = cbHi + (size_t)K * D_DIM;

  vq_c2_kernel<<<(K + 7) / 8, 256, 0, stream>>>(cb, c2, K);
  vq_cvt_cb_kernel<<<(K * D_DIM + 255) / 256, 256, 0, stream>>>(cb, cbHi, cbLo,
                                                                K * D_DIM);
  vq_argmin_kernel<<<N / M_TILE, 256, 0, stream>>>(x, cb, cbHi, c2, qout,
                                                   idx_out, K);
}
